// NaSwinAttention_50921132261467
// MI455X (gfx1250) — matrix operand order repaired and checked by reading, not compiled, not run on hardware
//
#include <hip/hip_runtime.h>
#include <hip/hip_bf16.h>

// ---------------------------------------------------------------------------
// Problem constants (match the reference)
// ---------------------------------------------------------------------------
constexpr int T = 8, HS = 64, WS = 64;
constexpr int WT = 4, WH = 8, WW = 8;
constexpr int NH = 12, HD = 128;
constexpr int VID_DIM = 1536, TXT_DIM = 1536, TXT_LEN = 256;
constexpr int L_VID   = T * HS * WS;                       // 32768
constexpr int N_WIN   = (T / WT) * (HS / WH) * (WS / WW);  // 128
constexpr int WIN_LEN = WT * WH * WW;                      // 256
constexpr int NHHD    = NH * HD;                           // 1536
constexpr int QKV_N   = 3 * NHHD;                          // 4608

typedef __bf16 bf16;
typedef __attribute__((ext_vector_type(16))) __bf16        v16bf;
typedef __attribute__((ext_vector_type(8)))  float         v8f;
typedef __attribute__((ext_vector_type(4)))  unsigned int  v4u;
typedef __attribute__((ext_vector_type(2)))  unsigned int  v2u;
typedef __attribute__((ext_vector_type(4)))  int           v4i;

union Frag {
    v16bf v;
    unsigned short s[16];
    unsigned int   u[8];
};

union Pack8 {  // 8 bf16 <-> one 128-bit register
    v4u  q;
    bf16 h[8];
};

__device__ __forceinline__ void lds_fence() {
    // order per-wave LDS stores before subsequent LDS loads (cross-lane data)
    asm volatile("s_wait_dscnt 0" ::: "memory");
}

// Optional CDNA5 async global->LDS copy (ASYNCcnt path); falls back cleanly.
#if __has_builtin(__builtin_amdgcn_global_load_async_to_lds_b128)
#define HAVE_ASYNC_LDS 1
typedef __attribute__((address_space(3))) v4i lds_v4i;
#else
#define HAVE_ASYNC_LDS 0
#endif

// ---------------------------------------------------------------------------
// f32 -> bf16 conversion (one pass; halves all subsequent HBM traffic)
// ---------------------------------------------------------------------------
__global__ __launch_bounds__(256) void f32_to_bf16_kernel(const float* __restrict__ in,
                                                          bf16* __restrict__ out,
                                                          size_t n) {
    size_t i = (size_t)blockIdx.x * blockDim.x + threadIdx.x;
    size_t stride = (size_t)gridDim.x * blockDim.x;
    for (; i < n; i += stride) out[i] = (bf16)in[i];
}

// ---------------------------------------------------------------------------
// Tiled bf16 WMMA GEMM: C[M,N] = A[M,K] @ B[K,N] + bias
// Block tile 128x128, K-step 32. 256 threads = 8 waves in a 4(M)x2(N) grid;
// each wave computes 32x64 = 2x4 C-tiles via v_wmma_f32_16x16x32_bf16.
// A staged naturally (A-frag = 2x ds_load_b128); B staged TRANSPOSED so each
// B-frag (lane = N-col, elements = K contiguous) is also 2x ds_load_b128.
// ---------------------------------------------------------------------------
template <bool OUT_F32>
__global__ __launch_bounds__(256) void gemm_bf16_kernel(const bf16* __restrict__ A,
                                                        const bf16* __restrict__ B,
                                                        const float* __restrict__ bias,
                                                        float* __restrict__ Cf,
                                                        bf16* __restrict__ Cb,
                                                        int M, int N, int K) {
    constexpr int BM = 128, BN = 128, BK = 32;
    constexpr int APAD = 40;  // row stride 80B -> conflict-free lane spread
    __shared__ bf16 As[BM][APAD];   // [m][k]
    __shared__ bf16 BsT[BN][APAD];  // [n][k]  (transposed tile)

    const int tid   = threadIdx.x;
    const int lane  = tid & 31;
    const int wave  = tid >> 5;      // 0..7
    const int waveM = wave >> 1;     // 0..3
    const int waveN = wave & 1;      // 0..1
    const int half  = lane >> 4;
    const int l16   = lane & 15;
    const int bm0   = blockIdx.y * BM;
    const int bn0   = blockIdx.x * BN;

    const int arow = tid >> 1, acol = (tid & 1) * 16;          // A staging map
    const int brow = tid >> 3, bcol = (tid & 7) * 16;          // B staging map
    const bf16* aSrc = A + (size_t)(bm0 + arow) * K + acol;
    const bf16* bSrc = B + (size_t)brow * N + bn0 + bcol;

    v8f acc[2][4];
#pragma unroll
    for (int i = 0; i < 2; ++i)
#pragma unroll
        for (int j = 0; j < 4; ++j) acc[i][j] = (v8f){};

    for (int k0 = 0; k0 < K; k0 += BK) {
        // ---- stage A tile (128x32): contiguous b128 copies ----
#if HAVE_ASYNC_LDS
        __builtin_amdgcn_global_load_async_to_lds_b128(
            (v4i*)(aSrc + k0), (lds_v4i*)&As[arow][acol], 0, 0);
#else
        {
            Pack8 a0; a0.q = *(const v4u*)(aSrc + k0);
            *(v4u*)&As[arow][acol] = a0.q;
        }
#endif
        // ---- stage B tile (32x128) transposed into BsT[n][k] ----
        {
            Pack8 b0, b1;
            const bf16* s = bSrc + (size_t)k0 * N;
            b0.q = *(const v4u*)s;
            b1.q = *(const v4u*)(s + 8);
#pragma unroll
            for (int j = 0; j < 8; ++j) {
                BsT[bcol + j][brow]     = b0.h[j];
                BsT[bcol + 8 + j][brow] = b1.h[j];
            }
        }
        // prefetch next K-step tiles while this one is consumed
        if (k0 + BK < K) {
            __builtin_prefetch(aSrc + k0 + BK, 0, 1);
            __builtin_prefetch(bSrc + (size_t)(k0 + BK) * N, 0, 1);
        }
#if HAVE_ASYNC_LDS
        asm volatile("s_wait_asynccnt 0" ::: "memory");
#endif
        __syncthreads();

        // ---- fragments: 2x b128 LDS loads each ----
        Frag afr[2], bfr[4];
#pragma unroll
        for (int mt = 0; mt < 2; ++mt) {
            int row = waveM * 32 + mt * 16 + l16;
            *(v4u*)&afr[mt].u[0] = *(const v4u*)&As[row][half * 8];       // K=half*8..+7
            *(v4u*)&afr[mt].u[4] = *(const v4u*)&As[row][16 + half * 8];  // K=16+half*8..+7
        }
#pragma unroll
        for (int nt = 0; nt < 4; ++nt) {
            int col = waveN * 64 + nt * 16 + l16;
            *(v4u*)&bfr[nt].u[0] = *(const v4u*)&BsT[col][half * 8];       // K=half*8..+7
            *(v4u*)&bfr[nt].u[4] = *(const v4u*)&BsT[col][16 + half * 8];  // K=16+half*8..+7
        }
#pragma unroll
        for (int mt = 0; mt < 2; ++mt)
#pragma unroll
            for (int nt = 0; nt < 4; ++nt)
                acc[mt][nt] = __builtin_amdgcn_wmma_f32_16x16x32_bf16(
                    false, afr[mt].v, false, bfr[nt].v, (short)0, acc[mt][nt], false, false);
        __syncthreads();
    }

    // ---- epilogue: C layout row = v + 8*half, col = l16 ----
#pragma unroll
    for (int mt = 0; mt < 2; ++mt) {
#pragma unroll
        for (int nt = 0; nt < 4; ++nt) {
            int col = bn0 + waveN * 64 + nt * 16 + l16;
            float bv = bias[col];
#pragma unroll
            for (int v = 0; v < 8; ++v) {
                int row = bm0 + waveM * 32 + mt * 16 + v + 8 * half;
                float val = acc[mt][nt][v] + bv;
                if (OUT_F32) Cf[(size_t)row * N + col] = val;
                else         Cb[(size_t)row * N + col] = (bf16)val;
            }
        }
    }
}

// ---------------------------------------------------------------------------
// RMSNorm on Q and K slices of qkv [L, 3, NH, HD] (bf16, in place).
// One wave per (token, head, q|k); b64 vector loads + wave32 shuffle reduce.
// ---------------------------------------------------------------------------
__global__ __launch_bounds__(256) void rmsnorm_qk_kernel(bf16* __restrict__ qkv,
                                                         const float* __restrict__ qscale,
                                                         const float* __restrict__ kscale,
                                                         int L) {
    int item = blockIdx.x * 8 + (threadIdx.x >> 5);
    int lane = threadIdx.x & 31;
    if (item >= L * NH * 2) return;
    int which = item & 1;                 // 0 = q, 1 = k
    int head  = (item >> 1) % NH;
    int tok   = (item >> 1) / NH;
    bf16* p = qkv + ((size_t)tok * 3 + which) * NHHD + (size_t)head * HD + lane * 4;
    union { v2u q; bf16 h[4]; } d;
    d.q = *(const v2u*)p;
    float x[4], ss = 0.0f;
#pragma unroll
    for (int i = 0; i < 4; ++i) { x[i] = (float)d.h[i]; ss += x[i] * x[i]; }
#pragma unroll
    for (int off = 16; off; off >>= 1) ss += __shfl_xor(ss, off, 32);
    float r = rsqrtf(ss * (1.0f / HD) + 1e-6f);
    const float* sc = (which ? kscale : qscale) + lane * 4;
#pragma unroll
    for (int i = 0; i < 4; ++i) d.h[i] = (bf16)(x[i] * r * sc[i]);
    *(v2u*)p = d.q;
}

// ---------------------------------------------------------------------------
// Flash attention per (window, head). 16 waves x 16 query rows = 256 queries.
// Per 32-key chunk: block-cooperative staging of K (natural) and V
// (transposed) into LDS, then WMMA S = Q K^T, online softmax in the WMMA C
// layout, P transposed through a per-wave LDS tile, WMMA O += P V.
// ---------------------------------------------------------------------------
template <bool IS_VID>
__device__ __forceinline__ int win_token(int w, int q) {
    if (!IS_VID) return q;
    int tb = w >> 6, hb = (w >> 3) & 7, wb = w & 7;
    int wt = q >> 6, wh = (q >> 3) & 7, ww = q & 7;
    return ((tb * 4 + wt) * 64 + hb * 8 + wh) * 64 + wb * 8 + ww;
}

template <bool IS_VID>
__global__ __launch_bounds__(512) void flash_attn_kernel(const bf16* __restrict__ qkv,
                                                         bf16* __restrict__ out) {
    __shared__ bf16 Ks[32][136];      // [key][d]   8.5 KB (row 272B, conflict-free)
    __shared__ bf16 VsT[128][40];     // [d][key]  10.0 KB (row 80B, conflict-free)
    __shared__ bf16 Pl[16][16][40];   // per-wave P tile, 20 KB

    const int w    = blockIdx.x;
    const int head = blockIdx.y;
    const int tid  = threadIdx.x;
    const int lane = tid & 31;
    const int wave = tid >> 5;   // 0..15
    const int half = lane >> 4;
    const int l16  = lane & 15;
    const int qbase = wave * 16;
    const float scale = 0.08838834764831845f;  // 1/sqrt(128)

    // ---- preload Q fragments (ISA A layout, 4 d-chunks of 32) ----
    const int qtok = win_token<IS_VID>(w, qbase + l16);
    const bf16* qb = qkv + ((size_t)qtok * 3 + 0) * NHHD + (size_t)head * HD;
    Frag qf[4];
#pragma unroll
    for (int kt = 0; kt < 4; ++kt) {
        *(v4u*)&qf[kt].u[0] = *(const v4u*)&qb[kt * 32 + half * 8];
        *(v4u*)&qf[kt].u[4] = *(const v4u*)&qb[kt * 32 + 16 + half * 8];
    }

    v8f o[8];
#pragma unroll
    for (int nt = 0; nt < 8; ++nt) o[nt] = (v8f){};
    float m[8], lsum[8];
#pragma unroll
    for (int v = 0; v < 8; ++v) { m[v] = -1e30f; lsum[v] = 0.0f; }

    const int skey = tid >> 4;        // 0..31: staged key row
    const int sdB  = (tid & 15) * 8;  // 0..120: staged d base

    for (int c = 0; c < WIN_LEN / 32; ++c) {
        // ---- cooperative staging of this chunk's K and V (1x global read) ----
        {
            int ktok = win_token<IS_VID>(w, c * 32 + skey);
            const bf16* kg = qkv + ((size_t)ktok * 3 + 1) * NHHD + (size_t)head * HD + sdB;
            *(v4u*)&Ks[skey][sdB] = *(const v4u*)kg;
            const bf16* vg = qkv + ((size_t)ktok * 3 + 2) * NHHD + (size_t)head * HD + sdB;
            Pack8 vv; vv.q = *(const v4u*)vg;
#pragma unroll
            for (int j = 0; j < 8; ++j) VsT[sdB + j][skey] = vv.h[j];
        }
        __syncthreads();

        // ---- S = Q K^T (2 key-tiles x 4 d-steps) ----
        v8f s[2];
#pragma unroll
        for (int nt = 0; nt < 2; ++nt) {
            int key = nt * 16 + l16;
            s[nt] = (v8f){};
#pragma unroll
            for (int kt = 0; kt < 4; ++kt) {
                Frag kf;
                *(v4u*)&kf.u[0] = *(const v4u*)&Ks[key][kt * 32 + half * 8];
                *(v4u*)&kf.u[4] = *(const v4u*)&Ks[key][kt * 32 + 16 + half * 8];
                s[nt] = __builtin_amdgcn_wmma_f32_16x16x32_bf16(
                    false, qf[kt].v, false, kf.v, (short)0, s[nt], false, false);
            }
        }

        // ---- online softmax (row = v + 8*half; 16-lane reductions) ----
        float p0v[8], p1v[8], corr[8];
#pragma unroll
        for (int v = 0; v < 8; ++v) {
            float s0 = s[0][v] * scale, s1 = s[1][v] * scale;
            float rm = fmaxf(s0, s1);
#pragma unroll
            for (int off = 8; off; off >>= 1) rm = fmaxf(rm, __shfl_xor(rm, off, 16));
            float mnew = fmaxf(m[v], rm);
            corr[v] = __expf(m[v] - mnew);
            float p0 = __expf(s0 - mnew), p1 = __expf(s1 - mnew);
            float rs = p0 + p1;
#pragma unroll
            for (int off = 8; off; off >>= 1) rs += __shfl_xor(rs, off, 16);
            lsum[v] = lsum[v] * corr[v] + rs;
            m[v] = mnew;
            p0v[v] = p0; p1v[v] = p1;
        }
#pragma unroll
        for (int nt = 0; nt < 8; ++nt)
#pragma unroll
            for (int v = 0; v < 8; ++v) o[nt][v] *= corr[v];

        // ---- P: C layout -> A layout via per-wave LDS tile ----
#pragma unroll
        for (int v = 0; v < 8; ++v) {
            Pl[wave][v + 8 * half][l16]      = (bf16)p0v[v];
            Pl[wave][v + 8 * half][16 + l16] = (bf16)p1v[v];
        }
        lds_fence();
        Frag pf;
        *(v4u*)&pf.u[0] = *(const v4u*)&Pl[wave][l16][half * 8];
        *(v4u*)&pf.u[4] = *(const v4u*)&Pl[wave][l16][16 + half * 8];

        // ---- O += P V (8 d-tiles; V-frag = 2x b128 from transposed LDS) ----
#pragma unroll
        for (int nt = 0; nt < 8; ++nt) {
            Frag vf;
            *(v4u*)&vf.u[0] = *(const v4u*)&VsT[nt * 16 + l16][half * 8];
            *(v4u*)&vf.u[4] = *(const v4u*)&VsT[nt * 16 + l16][16 + half * 8];
            o[nt] = __builtin_amdgcn_wmma_f32_16x16x32_bf16(
                false, pf.v, false, vf.v, (short)0, o[nt], false, false);
        }
        __syncthreads();  // protect Ks/VsT before next chunk's staging
    }

    // ---- normalize and write (window-reverse mapping folded in) ----
#pragma unroll
    for (int v = 0; v < 8; ++v) {
        int qrow = qbase + v + 8 * half;
        int tok  = win_token<IS_VID>(w, qrow);
        float inv = 1.0f / lsum[v];
#pragma unroll
        for (int nt = 0; nt < 8; ++nt)
            out[(size_t)tok * NHHD + (size_t)head * HD + nt * 16 + l16] = (bf16)(o[nt][v] * inv);
    }
}

// ---------------------------------------------------------------------------
// Host-side orchestration
// ---------------------------------------------------------------------------
static inline void convert(const float* in, bf16* out, size_t n, hipStream_t s) {
    int blocks = (int)((n + 255 * 4) / (256 * 4));
    if (blocks > 32768) blocks = 32768;
    f32_to_bf16_kernel<<<blocks, 256, 0, s>>>(in, out, n);
}

extern "C" void kernel_launch(void* const* d_in, const int* in_sizes, int n_in,
                              void* d_out, int out_size, void* d_ws, size_t ws_size,
                              hipStream_t stream) {
    const float* vid      = (const float*)d_in[0];
    const float* txt      = (const float*)d_in[1];
    const float* Wqkv_vid = (const float*)d_in[2];
    const float* bqkv_vid = (const float*)d_in[3];
    const float* Wqkv_txt = (const float*)d_in[4];
    const float* bqkv_txt = (const float*)d_in[5];
    const float* q_sc_vid = (const float*)d_in[6];
    const float* k_sc_vid = (const float*)d_in[7];
    const float* q_sc_txt = (const float*)d_in[8];
    const float* k_sc_txt = (const float*)d_in[9];
    const float* Wo_vid   = (const float*)d_in[10];
    const float* bo_vid   = (const float*)d_in[11];
    const float* Wo_txt   = (const float*)d_in[12];
    const float* bo_txt   = (const float*)d_in[13];

    float* vid_out = (float*)d_out;                              // [32768, 1536]
    float* txt_out = (float*)d_out + (size_t)L_VID * VID_DIM;    // [256, 1536]

    // workspace carve-up (bf16 elements)
    bf16* ws = (bf16*)d_ws;
    size_t off = 0;
    bf16* vidB  = ws + off; off += (size_t)L_VID * VID_DIM;
    bf16* txtB  = ws + off; off += (size_t)TXT_LEN * TXT_DIM;
    bf16* WqvB  = ws + off; off += (size_t)VID_DIM * QKV_N;
    bf16* WqtB  = ws + off; off += (size_t)TXT_DIM * QKV_N;
    bf16* WovB  = ws + off; off += (size_t)NHHD * VID_DIM;
    bf16* WotB  = ws + off; off += (size_t)NHHD * TXT_DIM;
    bf16* qkvV  = ws + off; off += (size_t)L_VID * QKV_N;
    bf16* qkvT  = ws + off; off += (size_t)TXT_LEN * QKV_N;
    bf16* attV  = ws + off; off += (size_t)L_VID * NHHD;
    bf16* attT  = ws + off; off += (size_t)TXT_LEN * NHHD;

    // 1) down-convert activations + weights to bf16 (single pass each)
    convert(vid,      vidB, (size_t)L_VID * VID_DIM,   stream);
    convert(txt,      txtB, (size_t)TXT_LEN * TXT_DIM, stream);
    convert(Wqkv_vid, WqvB, (size_t)VID_DIM * QKV_N,   stream);
    convert(Wqkv_txt, WqtB, (size_t)TXT_DIM * QKV_N,   stream);
    convert(Wo_vid,   WovB, (size_t)NHHD * VID_DIM,    stream);
    convert(Wo_txt,   WotB, (size_t)NHHD * TXT_DIM,    stream);

    // 2) QKV projections (WMMA GEMM, bf16 out)
    gemm_bf16_kernel<false><<<dim3(QKV_N / 128, L_VID / 128), 256, 0, stream>>>(
        vidB, WqvB, bqkv_vid, nullptr, qkvV, L_VID, QKV_N, VID_DIM);
    gemm_bf16_kernel<false><<<dim3(QKV_N / 128, TXT_LEN / 128), 256, 0, stream>>>(
        txtB, WqtB, bqkv_txt, nullptr, qkvT, TXT_LEN, QKV_N, TXT_DIM);

    // 3) RMSNorm on Q and K (in place)
    rmsnorm_qk_kernel<<<(L_VID * NH * 2) / 8, 256, 0, stream>>>(qkvV, q_sc_vid, k_sc_vid, L_VID);
    rmsnorm_qk_kernel<<<(TXT_LEN * NH * 2) / 8, 256, 0, stream>>>(qkvT, q_sc_txt, k_sc_txt, TXT_LEN);

    // 4) flash attention (windowed for vid, global 256-token for txt)
    flash_attn_kernel<true><<<dim3(N_WIN, NH), 512, 0, stream>>>(qkvV, attV);
    flash_attn_kernel<false><<<dim3(1, NH), 512, 0, stream>>>(qkvT, attT);

    // 5) output projections (WMMA GEMM, f32 out straight into d_out)
    gemm_bf16_kernel<true><<<dim3(VID_DIM / 128, L_VID / 128), 256, 0, stream>>>(
        attV, WovB, bo_vid, vid_out, nullptr, L_VID, VID_DIM, NHHD);
    gemm_bf16_kernel<true><<<dim3(TXT_DIM / 128, TXT_LEN / 128), 256, 0, stream>>>(
        attT, WotB, bo_txt, txt_out, nullptr, TXT_LEN, TXT_DIM, NHHD);
}
